// Multihead_GRUcell_5446018531354
// MI455X (gfx1250) — matrix operand order repaired and checked by reading, not compiled, not run on hardware
//
#include <hip/hip_runtime.h>

#define BB 16384
#define HH 8
#define DD 256
#define HDIM 128
#define KK 384          // HDIM + DD
#define TM 128          // batch rows per workgroup
#define AROW 136        // padded A-buffer row stride (ushorts), 16B aligned

typedef __attribute__((ext_vector_type(16))) __bf16 v16bf;
typedef __attribute__((ext_vector_type(8)))  float  v8f;
typedef __attribute__((ext_vector_type(4)))  unsigned int uint4v;

__device__ __forceinline__ unsigned short f2bf(float f) {
    unsigned u = __builtin_bit_cast(unsigned, f);
    unsigned r = u + 0x7FFFu + ((u >> 16) & 1u);   // round-to-nearest-even
    return (unsigned short)(r >> 16);
}
__device__ __forceinline__ float bf2f(unsigned short s) {
    unsigned u = ((unsigned)s) << 16;
    return __builtin_bit_cast(float, u);
}
#if __has_builtin(__builtin_amdgcn_cvt_pk_bf16_f32)
__device__ __forceinline__ unsigned pack2(float a, float b) {
    return __builtin_bit_cast(unsigned, __builtin_amdgcn_cvt_pk_bf16_f32(a, b));
}
#else
__device__ __forceinline__ unsigned pack2(float a, float b) {
    return (unsigned)f2bf(a) | ((unsigned)f2bf(b) << 16);
}
#endif
__device__ __forceinline__ v16bf mk_frag(uint4v lo, uint4v hi) {
    return __builtin_bit_cast(v16bf, __builtin_shufflevector(lo, hi, 0,1,2,3,4,5,6,7));
}

__global__ __launch_bounds__(256)
void gru_fused_kernel(const float* __restrict__ x, const float* __restrict__ h,
                      const float* __restrict__ sw, const float* __restrict__ sb,
                      const float* __restrict__ tw, const float* __restrict__ tb,
                      const float* __restrict__ ow, const float* __restrict__ ob,
                      float* __restrict__ out, float* __restrict__ nh_out) {
    extern __shared__ char smem_raw[];
    unsigned short* lds_a = (unsigned short*)smem_raw;                       // TM x AROW bf16
    unsigned*       lds_w = (unsigned*)(smem_raw + TM * AROW * 2);           // 24576 dwords (B frags)
    float*          lds_sig = (float*)(smem_raw + TM * AROW * 2 + 24576 * 4);// 2*TM

    const int tid  = threadIdx.x;
    const int wave = tid >> 5;
    const int lane = tid & 31;
    const int head = blockIdx.y;
    const int row0 = blockIdx.x * TM;

    // Prefetch this head's out_w slice (used in Phase E) into cache early.
    {
        const char* owp = (const char*)(ow + (size_t)head * HDIM * DD);
        for (int i = tid; i < 1024; i += 256)               // 1024 x 128B lines = 128KB
            __builtin_prefetch(owp + (size_t)i * 128, 0, 1);
    }

    // ---------------- Phase A: stage bf16(h tile) + tanh_w (B-fragment layout) ----------------
    for (int i = tid; i < TM * (HDIM / 4); i += 256) {
        int r = i / (HDIM / 4), c4 = i % (HDIM / 4);
        const float4 v = ((const float4*)(h + ((size_t)(row0 + r) * HH + head) * HDIM))[c4];
        unsigned* dst = (unsigned*)(lds_a + r * AROW + c4 * 4);
        dst[0] = pack2(v.x, v.y);
        dst[1] = pack2(v.z, v.w);
    }
    const float* twh = tw + (size_t)head * KK * HDIM;
    for (int w = tid; w < 12 * 8 * 32 * 8; w += 256) {   // kt=12, nt=8, 32 lanes, 8 dwords
        int v  = w & 7;
        int ln = (w >> 3) & 31;
        int nt = (w >> 8) & 7;
        int kt = w >> 11;
        int k0 = kt * 32 + (v >> 2) * 16 + (ln >> 4) * 8 + 2 * (v & 3);
        int n  = nt * 16 + (ln & 15);
        lds_w[w] = pack2(twh[(size_t)k0 * HDIM + n], twh[(size_t)(k0 + 1) * HDIM + n]);
    }
    __syncthreads();

    // ---------------- Phase B: sigmoid gates (N=2, VALU) ----------------
    if (tid < TM) {
        int r = tid;
        const float* swh = sw + (size_t)head * KK * 2;
        float a0 = sb[head * 2 + 0], a1 = sb[head * 2 + 1];
        for (int k = 0; k < HDIM; ++k) {
            float v = bf2f(lds_a[r * AROW + k]);
            a0 += v * swh[k * 2 + 0];
            a1 += v * swh[k * 2 + 1];
        }
        const float* xr = x + (size_t)(row0 + r) * DD;
        for (int k = 0; k < DD; ++k) {
            float v = xr[k];
            a0 += v * swh[(HDIM + k) * 2 + 0];
            a1 += v * swh[(HDIM + k) * 2 + 1];
        }
        lds_sig[r * 2 + 0] = 1.f / (1.f + __expf(-a0));
        lds_sig[r * 2 + 1] = 1.f / (1.f + __expf(-a1));
    }
    __syncthreads();

    // ---------------- Phase C: tanh_in[:, :128] = sig0 * h (in place) ----------------
    for (int i = tid; i < TM * (HDIM / 2); i += 256) {
        int r = i / (HDIM / 2), c = i % (HDIM / 2);
        unsigned* p = (unsigned*)(lds_a + r * AROW) + c;
        unsigned old = *p;
        float s0 = lds_sig[r * 2];
        *p = pack2(s0 * bf2f((unsigned short)(old & 0xffff)),
                   s0 * bf2f((unsigned short)(old >> 16)));
    }
    __syncthreads();

    // ---------------- Phase D: tanh GEMM (128x384 @ 384x128, bf16 WMMA) ----------------
    const int mrow  = lane & 15;      // A row within 16-row slice
    const int khalf = lane >> 4;      // K half select for A frags
    const int ncol  = lane & 15;      // C column within n-tile
    const int mhalf = lane >> 4;      // C row half
    const int wrow  = wave * 16 + mrow;

    v8f acc[8] = {};
#pragma unroll
    for (int kt = 0; kt < 12; ++kt) {
        uint4v lo, hi;
        if (kt < 4) {
            const unsigned short* ap = lds_a + wrow * AROW + kt * 32 + khalf * 8;
            lo = *(const uint4v*)ap;
            hi = *(const uint4v*)(ap + 16);
        } else {
            const float4* xp = (const float4*)(x + (size_t)(row0 + wrow) * DD + (kt - 4) * 32 + khalf * 8);
            float4 p0 = xp[0], p1 = xp[1], p2 = xp[4], p3 = xp[5];
            lo = (uint4v){pack2(p0.x, p0.y), pack2(p0.z, p0.w), pack2(p1.x, p1.y), pack2(p1.z, p1.w)};
            hi = (uint4v){pack2(p2.x, p2.y), pack2(p2.z, p2.w), pack2(p3.x, p3.y), pack2(p3.z, p3.w)};
        }
        v16bf afrag = mk_frag(lo, hi);
#pragma unroll
        for (int nt = 0; nt < 8; ++nt) {
            const uint4v* bp = (const uint4v*)(lds_w + ((kt * 8 + nt) * 32 + lane) * 8);
            v16bf bfrag = mk_frag(bp[0], bp[1]);
            acc[nt] = __builtin_amdgcn_wmma_f32_16x16x32_bf16(
                false, afrag, false, bfrag, (short)0, acc[nt], false, false);
        }
    }

    // Epilogue: cand=tanh(.), new_h=(1-p)*h+p*cand ; write new_h (f32) + bf16 into A buffer
#pragma unroll
    for (int nt = 0; nt < 8; ++nt) {
#pragma unroll
        for (int j = 0; j < 8; ++j) {
            int trow = wave * 16 + j + mhalf * 8;
            int grow = row0 + trow;
            int n    = nt * 16 + ncol;
            float cand = tanhf(acc[nt][j] + tb[head * HDIM + n]);
            float p    = lds_sig[trow * 2 + 1];
            float hv   = h[((size_t)grow * HH + head) * HDIM + n];
            float nhv  = (1.f - p) * hv + p * cand;
            nh_out[((size_t)grow * HH + head) * HDIM + n] = nhv;
            lds_a[trow * AROW + n] = f2bf(nhv);
        }
    }
    __syncthreads();

    // ---------------- Phase E: restage out_w into B-fragment layout ----------------
    const float* owh = ow + (size_t)head * HDIM * DD;
    for (int w = tid; w < 4 * 16 * 32 * 8; w += 256) {   // kt=4, nt=16
        int v  = w & 7;
        int ln = (w >> 3) & 31;
        int nt = (w >> 8) & 15;
        int kt = w >> 12;
        int k0 = kt * 32 + (v >> 2) * 16 + (ln >> 4) * 8 + 2 * (v & 3);
        int n  = nt * 16 + (ln & 15);
        lds_w[w] = pack2(owh[(size_t)k0 * DD + n], owh[(size_t)(k0 + 1) * DD + n]);
    }
    __syncthreads();

    // ---------------- Phase F: output GEMM (128x128 @ 128x256), two N-halves ----------------
    // Split into halves of 8 n-tiles each to keep accumulator VGPR pressure < 256.
#pragma unroll
    for (int half = 0; half < 2; ++half) {
        v8f acc2[8] = {};
#pragma unroll
        for (int kt = 0; kt < 4; ++kt) {
            const unsigned short* ap = lds_a + wrow * AROW + kt * 32 + khalf * 8;
            v16bf afrag = mk_frag(*(const uint4v*)ap, *(const uint4v*)(ap + 16));
#pragma unroll
            for (int nt = 0; nt < 8; ++nt) {
                int ntg = half * 8 + nt;
                const uint4v* bp = (const uint4v*)(lds_w + ((kt * 16 + ntg) * 32 + lane) * 8);
                v16bf bfrag = mk_frag(bp[0], bp[1]);
                acc2[nt] = __builtin_amdgcn_wmma_f32_16x16x32_bf16(
                    false, afrag, false, bfrag, (short)0, acc2[nt], false, false);
            }
        }
#pragma unroll
        for (int nt = 0; nt < 8; ++nt) {
#pragma unroll
            for (int j = 0; j < 8; ++j) {
                int grow = row0 + wave * 16 + j + mhalf * 8;
                int n    = (half * 8 + nt) * 16 + ncol;
                out[((size_t)grow * HH + head) * DD + n] = acc2[nt][j] + ob[head * DD + n];
            }
        }
    }
}

extern "C" void kernel_launch(void* const* d_in, const int* in_sizes, int n_in,
                              void* d_out, int out_size, void* d_ws, size_t ws_size,
                              hipStream_t stream) {
    (void)in_sizes; (void)n_in; (void)out_size; (void)d_ws; (void)ws_size;
    const float* x  = (const float*)d_in[0];
    const float* h  = (const float*)d_in[1];
    const float* sw = (const float*)d_in[2];
    const float* sb = (const float*)d_in[3];
    const float* tw = (const float*)d_in[4];
    const float* tb = (const float*)d_in[5];
    const float* ow = (const float*)d_in[6];
    const float* ob = (const float*)d_in[7];
    float* out = (float*)d_out;
    float* nh  = out + (size_t)BB * HH * DD;   // tuple: (output, new_h) concatenated

    dim3 grid(BB / TM, HH);
    size_t smem = (size_t)TM * AROW * 2   // A buffer (bf16)
                + (size_t)24576 * 4       // weight B-fragment buffer
                + (size_t)2 * TM * 4;     // sigmoid gates
    gru_fused_kernel<<<grid, 256, smem, stream>>>(x, h, sw, sb, tw, tb, ow, ob, out, nh);
}
